// LRU_80032420594092
// MI455X (gfx1250) — matrix operand order repaired and checked by reading, not compiled, not run on hardware
//
#include <hip/hip_runtime.h>
#include <hip/hip_bf16.h>

// ---------------------------------------------------------------------------
// LRU forward for MI455X (gfx1250): f16 WMMA GEMMs (4x1 register-blocked) +
// chunked f32 complex scan with interleaved complex Bu (b64 traffic).
// ---------------------------------------------------------------------------
#define T_LEN   32768
#define DM      512          // d_model
#define DH      512          // d_hidden
#define CHUNK_T 256
#define NCHUNK  (T_LEN / CHUNK_T)   // 128

typedef __attribute__((ext_vector_type(16))) _Float16 v16h;
typedef __attribute__((ext_vector_type(8)))  _Float16 v8h;
typedef __attribute__((ext_vector_type(4)))  _Float16 v4h;
typedef __attribute__((ext_vector_type(8)))  float    v8f;

union U16 { v16h v; v8h h[2]; };

// A fragment, 16x32 f16 (ISA 7.12.2): lane (l,hi) holds row M=l,
// halves 0..7 = K[hi*8 .. hi*8+7], halves 8..15 = K[16+hi*8 .. +7].
__device__ __forceinline__ v16h load_frag_A(const _Float16* p) {
  U16 u;
  u.h[0] = *(const v8h*)(p);
  u.h[1] = *(const v8h*)(p + 16);
  return u.v;
}

// B fragment, 32x16 f16: lane (l,hi) holds column N=l, K = hi*8 + 0..7, 16 + hi*8 + 0..7.
__device__ __forceinline__ v16h load_frag_B(const _Float16* p) {
  U16 u;
  u.h[0] = *(const v8h*)(p);
  u.h[1] = *(const v8h*)(p + 16);
  return u.v;
}

#define WMMA_F16(a, b, c) \
  __builtin_amdgcn_wmma_f32_16x16x32_f16(false, (a), false, (b), (short)0, (c), false, false)

// --------------------------- precompute kernels ----------------------------

// lambda = exp(-exp(nu)) * exp(i*exp(theta)) ; lam[0..511]=re, lam[512..]=im
__global__ void lru_lambda(const float* __restrict__ theta_log,
                           const float* __restrict__ nu_log,
                           float* __restrict__ lam) {
  int h = blockIdx.x * 256 + threadIdx.x;
  float r  = expf(-expf(nu_log[h]));
  float ph = expf(theta_log[h]);
  lam[h]      = r * cosf(ph);
  lam[DH + h] = r * sinf(ph);
}

// B_norm = (B_re + i B_im) * exp(gamma)[h], converted to f16 (row-major HxD)
__global__ void lru_conv_b(const float* __restrict__ B_re,
                           const float* __restrict__ B_im,
                           const float* __restrict__ gamma_log,
                           _Float16* __restrict__ breh,
                           _Float16* __restrict__ bimh) {
  int i = blockIdx.x * 256 + threadIdx.x;   // covers DH*DM
  int h = i >> 9;
  float g = expf(gamma_log[h]);
  breh[i] = (_Float16)(B_re[i] * g);
  bimh[i] = (_Float16)(B_im[i] * g);
}

// C_re -> f16 ; C_im -> -C_im f16 (sign folded so GEMM2 is pure accumulate)
__global__ void lru_conv_c(const float* __restrict__ C_re,
                           const float* __restrict__ C_im,
                           _Float16* __restrict__ creh,
                           _Float16* __restrict__ cnimh) {
  int i = blockIdx.x * 256 + threadIdx.x;   // covers DM*DH
  creh[i]  = (_Float16)(C_re[i]);
  cnimh[i] = (_Float16)(-C_im[i]);
}

// x (f32) -> f16, vectorized 4-wide
__global__ void lru_conv_x(const float* __restrict__ x,
                           _Float16* __restrict__ xh) {
  size_t i = ((size_t)blockIdx.x * 256 + threadIdx.x) * 4;
  float4 f = *(const float4*)(x + i);
  v4h h;
  h[0] = (_Float16)f.x; h[1] = (_Float16)f.y;
  h[2] = (_Float16)f.z; h[3] = (_Float16)f.w;
  *(v4h*)(xh + i) = h;
}

// ------------------------------- GEMM 1 ------------------------------------
// Bu (T x DH, interleaved complex f32) = x_h @ Bnorm^T.
// Register blocking: each wave owns 4 M-tiles x 1 N-tile; B fragments
// (re and im) are loaded once per K-step and shared across 4 A fragments
// -> 12 b128 loads per 8 WMMAs.
__global__ void lru_gemm1(const _Float16* __restrict__ xh,
                          const _Float16* __restrict__ breh,
                          const _Float16* __restrict__ bimh,
                          float2* __restrict__ bu) {
  const int lane = threadIdx.x & 31;
  const int wave = threadIdx.x >> 5;
  const int tile = blockIdx.x * 8 + wave;   // 16384 wave-tiles
  const int nt = tile & 31;                 // DH/16 = 32
  const int mq = tile >> 5;                 // (T/64) = 512 macro-M tiles
  const int l  = lane & 15;
  const int hi = lane >> 4;

  const _Float16* a0 = xh + (size_t)(mq * 64 +  0 + l) * DM + hi * 8;
  const _Float16* a1 = xh + (size_t)(mq * 64 + 16 + l) * DM + hi * 8;
  const _Float16* a2 = xh + (size_t)(mq * 64 + 32 + l) * DM + hi * 8;
  const _Float16* a3 = xh + (size_t)(mq * 64 + 48 + l) * DM + hi * 8;
  const _Float16* br = breh + (size_t)(nt * 16 + l) * DM + hi * 8;
  const _Float16* bi = bimh + (size_t)(nt * 16 + l) * DM + hi * 8;

  v8f cre0 = {}, cre1 = {}, cre2 = {}, cre3 = {};
  v8f cim0 = {}, cim1 = {}, cim2 = {}, cim3 = {};
#pragma unroll 2
  for (int k0 = 0; k0 < DM; k0 += 32) {
    v16h fb0 = load_frag_B(br + k0);
    v16h fb1 = load_frag_B(bi + k0);
    v16h fa0 = load_frag_A(a0 + k0);
    v16h fa1 = load_frag_A(a1 + k0);
    v16h fa2 = load_frag_A(a2 + k0);
    v16h fa3 = load_frag_A(a3 + k0);
    cre0 = WMMA_F16(fa0, fb0, cre0);  cim0 = WMMA_F16(fa0, fb1, cim0);
    cre1 = WMMA_F16(fa1, fb0, cre1);  cim1 = WMMA_F16(fa1, fb1, cim1);
    cre2 = WMMA_F16(fa2, fb0, cre2);  cim2 = WMMA_F16(fa2, fb1, cim2);
    cre3 = WMMA_F16(fa3, fb0, cre3);  cim3 = WMMA_F16(fa3, fb1, cim3);
  }
  const int col = nt * 16 + l;
#pragma unroll
  for (int v = 0; v < 8; ++v) {             // C/D layout: M = hi*8 + v, N = l
    int r0 = mq * 64 + hi * 8 + v;
    bu[(size_t)(r0     ) * DH + col] = make_float2(cre0[v], cim0[v]);
    bu[(size_t)(r0 + 16) * DH + col] = make_float2(cre1[v], cim1[v]);
    bu[(size_t)(r0 + 32) * DH + col] = make_float2(cre2[v], cim2[v]);
    bu[(size_t)(r0 + 48) * DH + col] = make_float2(cre3[v], cim3[v]);
  }
}

// ------------------------------ chunked scan -------------------------------
// Phase 1: per (chunk, h) compute chunk transform s -> A*s + b (complex),
// with A zeroed by any reset inside the chunk.
__global__ void lru_scan_p1(const float2* __restrict__ bu,
                            const unsigned char* __restrict__ start,
                            const float* __restrict__ lam,
                            float* __restrict__ cA_re, float* __restrict__ cA_im,
                            float* __restrict__ cB_re, float* __restrict__ cB_im) {
  int h     = (blockIdx.x & 1) * 256 + threadIdx.x;
  int chunk = blockIdx.x >> 1;
  float lr = lam[h], li = lam[DH + h];
  float Ar = 1.f, Ai = 0.f, br = 0.f, bi = 0.f;
  int t0 = chunk * CHUNK_T;
  for (int i = 0; i < CHUNK_T; ++i) {
    int t = t0 + i;
    float keep = start[t] ? 0.f : 1.f;
    float ar = lr * keep, ai = li * keep;
    float2 u = bu[(size_t)t * DH + h];
    float nAr = ar * Ar - ai * Ai;
    float nAi = ar * Ai + ai * Ar;
    float nbr = ar * br - ai * bi + u.x;
    float nbi = ar * bi + ai * br + u.y;
    Ar = nAr; Ai = nAi; br = nbr; bi = nbi;
  }
  size_t o = (size_t)chunk * DH + h;
  cA_re[o] = Ar; cA_im[o] = Ai; cB_re[o] = br; cB_im[o] = bi;
}

// Phase 2: sequential prefix over the 128 chunk transforms per hidden dim;
// overwrites cA_* with each chunk's carry-in state.
__global__ void lru_scan_p2(const float* __restrict__ state,
                            float* cA_re, float* cA_im,
                            const float* __restrict__ cB_re,
                            const float* __restrict__ cB_im) {
  int h = blockIdx.x * 256 + threadIdx.x;
  float sr = state[h], si = 0.f;            // initial state is real
  for (int c = 0; c < NCHUNK; ++c) {
    size_t o = (size_t)c * DH + h;
    float Ar = cA_re[o], Ai = cA_im[o];
    float br = cB_re[o], bi = cB_im[o];
    cA_re[o] = sr; cA_im[o] = si;           // carry-in for chunk c
    float nr = Ar * sr - Ai * si + br;
    float ni = Ar * si + Ai * sr + bi;
    sr = nr; si = ni;
  }
}

// Phase 3: replay recurrence per chunk with carry-in; emit xs as f16 planes
// for GEMM2 and the final complex state (interleaved re,im) to d_out.
__global__ void lru_scan_p3(const float2* __restrict__ bu,
                            const unsigned char* __restrict__ start,
                            const float* __restrict__ lam,
                            const float* __restrict__ carry_re,
                            const float* __restrict__ carry_im,
                            _Float16* __restrict__ xsre,
                            _Float16* __restrict__ xsim,
                            float* __restrict__ out_state) {
  int h     = (blockIdx.x & 1) * 256 + threadIdx.x;
  int chunk = blockIdx.x >> 1;
  size_t o = (size_t)chunk * DH + h;
  float sr = carry_re[o], si = carry_im[o];
  float lr = lam[h], li = lam[DH + h];
  int t0 = chunk * CHUNK_T;
  for (int i = 0; i < CHUNK_T; ++i) {
    int t = t0 + i;
    float keep = start[t] ? 0.f : 1.f;
    float pr = sr * keep, pi = si * keep;
    size_t idx = (size_t)t * DH + h;
    float2 u = bu[idx];
    sr = lr * pr - li * pi + u.x;
    si = lr * pi + li * pr + u.y;
    xsre[idx] = (_Float16)sr;
    xsim[idx] = (_Float16)si;
  }
  if (chunk == NCHUNK - 1) {
    out_state[2 * h]     = sr;
    out_state[2 * h + 1] = si;
  }
}

// ------------------------------- GEMM 2 ------------------------------------
// y (T x DM) = xs_re @ C_re^T + xs_im @ (-C_im)^T + x * D
// Same 4x1 register blocking; C fragments shared across 4 M-tiles.
__global__ void lru_gemm2(const _Float16* __restrict__ xsre,
                          const _Float16* __restrict__ xsim,
                          const _Float16* __restrict__ creh,
                          const _Float16* __restrict__ cnimh,
                          const float* __restrict__ x,
                          const float* __restrict__ Dv,
                          float* __restrict__ y) {
  const int lane = threadIdx.x & 31;
  const int wave = threadIdx.x >> 5;
  const int tile = blockIdx.x * 8 + wave;
  const int nt = tile & 31;                 // DM/16 = 32
  const int mq = tile >> 5;                 // T/64 = 512
  const int l  = lane & 15;
  const int hi = lane >> 4;

  const _Float16* ar0 = xsre + (size_t)(mq * 64 +  0 + l) * DH + hi * 8;
  const _Float16* ar1 = xsre + (size_t)(mq * 64 + 16 + l) * DH + hi * 8;
  const _Float16* ar2 = xsre + (size_t)(mq * 64 + 32 + l) * DH + hi * 8;
  const _Float16* ar3 = xsre + (size_t)(mq * 64 + 48 + l) * DH + hi * 8;
  const _Float16* ai0 = xsim + (size_t)(mq * 64 +  0 + l) * DH + hi * 8;
  const _Float16* ai1 = xsim + (size_t)(mq * 64 + 16 + l) * DH + hi * 8;
  const _Float16* ai2 = xsim + (size_t)(mq * 64 + 32 + l) * DH + hi * 8;
  const _Float16* ai3 = xsim + (size_t)(mq * 64 + 48 + l) * DH + hi * 8;
  const _Float16* br  = creh  + (size_t)(nt * 16 + l) * DH + hi * 8;
  const _Float16* bn  = cnimh + (size_t)(nt * 16 + l) * DH + hi * 8;

  v8f acc0 = {}, acc1 = {}, acc2 = {}, acc3 = {};
#pragma unroll 2
  for (int k0 = 0; k0 < DH; k0 += 32) {
    v16h fbr = load_frag_B(br + k0);
    v16h fbn = load_frag_B(bn + k0);
    v16h f0  = load_frag_A(ar0 + k0);
    v16h f1  = load_frag_A(ar1 + k0);
    v16h f2  = load_frag_A(ar2 + k0);
    v16h f3  = load_frag_A(ar3 + k0);
    acc0 = WMMA_F16(f0, fbr, acc0);
    acc1 = WMMA_F16(f1, fbr, acc1);
    acc2 = WMMA_F16(f2, fbr, acc2);
    acc3 = WMMA_F16(f3, fbr, acc3);
    v16h g0 = load_frag_A(ai0 + k0);
    v16h g1 = load_frag_A(ai1 + k0);
    v16h g2 = load_frag_A(ai2 + k0);
    v16h g3 = load_frag_A(ai3 + k0);
    acc0 = WMMA_F16(g0, fbn, acc0);
    acc1 = WMMA_F16(g1, fbn, acc1);
    acc2 = WMMA_F16(g2, fbn, acc2);
    acc3 = WMMA_F16(g3, fbn, acc3);
  }
  const int col = nt * 16 + l;
  const float d = Dv[col];
#pragma unroll
  for (int v = 0; v < 8; ++v) {
    int r0 = mq * 64 + hi * 8 + v;
    size_t i0 = (size_t)(r0     ) * DM + col;
    size_t i1 = (size_t)(r0 + 16) * DM + col;
    size_t i2 = (size_t)(r0 + 32) * DM + col;
    size_t i3 = (size_t)(r0 + 48) * DM + col;
    y[i0] = acc0[v] + x[i0] * d;
    y[i1] = acc1[v] + x[i1] * d;
    y[i2] = acc2[v] + x[i2] * d;
    y[i3] = acc3[v] + x[i3] * d;
  }
}

// ------------------------------- launcher ----------------------------------
extern "C" void kernel_launch(void* const* d_in, const int* in_sizes, int n_in,
                              void* d_out, int out_size, void* d_ws, size_t ws_size,
                              hipStream_t stream) {
  (void)in_sizes; (void)n_in; (void)out_size; (void)ws_size;

  const float*         state     = (const float*)d_in[0];
  const float*         x         = (const float*)d_in[1];
  const unsigned char* start     = (const unsigned char*)d_in[2];
  const float*         theta_log = (const float*)d_in[3];
  const float*         nu_log    = (const float*)d_in[4];
  const float*         gamma_log = (const float*)d_in[5];
  const float*         B_re      = (const float*)d_in[6];
  const float*         B_im      = (const float*)d_in[7];
  const float*         C_re      = (const float*)d_in[8];
  const float*         C_im      = (const float*)d_in[9];
  const float*         Dv        = (const float*)d_in[10];

  float* out_state = (float*)d_out;          // 512 complex -> 1024 f32
  float* y         = (float*)d_out + 2 * DH; // T x DM f32

  char* w = (char*)d_ws;
  size_t off = 0;
  auto alloc = [&](size_t bytes) -> void* {
    void* p = w + off;
    off += (bytes + 255) & ~(size_t)255;
    return p;
  };
  _Float16* xh    = (_Float16*)alloc((size_t)T_LEN * DM * 2);      // 32 MB
  _Float16* breh  = (_Float16*)alloc((size_t)DH * DM * 2);
  _Float16* bimh  = (_Float16*)alloc((size_t)DH * DM * 2);
  _Float16* creh  = (_Float16*)alloc((size_t)DM * DH * 2);
  _Float16* cnimh = (_Float16*)alloc((size_t)DM * DH * 2);
  float*    lam   = (float*)alloc((size_t)2 * DH * 4);
  float2*   bu    = (float2*)alloc((size_t)T_LEN * DH * 8);        // 128 MB
  float*    cA_re = (float*)alloc((size_t)NCHUNK * DH * 4);
  float*    cA_im = (float*)alloc((size_t)NCHUNK * DH * 4);
  float*    cB_re = (float*)alloc((size_t)NCHUNK * DH * 4);
  float*    cB_im = (float*)alloc((size_t)NCHUNK * DH * 4);
  _Float16* xsre  = (_Float16*)alloc((size_t)T_LEN * DH * 2);      // 32 MB
  _Float16* xsim  = (_Float16*)alloc((size_t)T_LEN * DH * 2);      // 32 MB

  lru_lambda<<<2, 256, 0, stream>>>(theta_log, nu_log, lam);
  lru_conv_b<<<(DH * DM) / 256, 256, 0, stream>>>(B_re, B_im, gamma_log, breh, bimh);
  lru_conv_c<<<(DM * DH) / 256, 256, 0, stream>>>(C_re, C_im, creh, cnimh);
  lru_conv_x<<<(T_LEN * DM) / (256 * 4), 256, 0, stream>>>(x, xh);

  // 4x1 register-blocked WMMA GEMMs: (T/64)*(DH/16) wave-tiles, 8 waves/block
  lru_gemm1<<<(T_LEN / 64) * (DH / 16) / 8, 256, 0, stream>>>(xh, breh, bimh, bu);

  lru_scan_p1<<<NCHUNK * (DH / 256), 256, 0, stream>>>(bu, start, lam,
                                                       cA_re, cA_im, cB_re, cB_im);
  lru_scan_p2<<<DH / 256, 256, 0, stream>>>(state, cA_re, cA_im, cB_re, cB_im);
  lru_scan_p3<<<NCHUNK * (DH / 256), 256, 0, stream>>>(bu, start, lam,
                                                       cA_re, cA_im, xsre, xsim,
                                                       out_state);

  lru_gemm2<<<(T_LEN / 64) * (DM / 16) / 8, 256, 0, stream>>>(xsre, xsim, creh,
                                                              cnimh, x, Dv, y);
}
